// LoTD_55619826483669
// MI455X (gfx1250) — matrix operand order repaired and checked by reading, not compiled, not run on hardware
//
#include <hip/hip_runtime.h>

#define BS   32
#define NS   576
#define NT   576
#define HIDD 64
#define NTILES 36          // 576 / 16
#define NITERS 50
#define REGC   0.1f
#define INVREG 10.0f
#define LOGA  (-6.35610766069589f)   // -log(576)

typedef __attribute__((ext_vector_type(16))) __bf16 v16bf;
typedef __attribute__((ext_vector_type(8)))  __bf16 v8bf;
typedef __attribute__((ext_vector_type(8)))  float  v8f;
typedef unsigned int u32;
typedef u32 __attribute__((ext_vector_type(4))) v4u32;
typedef int __attribute__((ext_vector_type(4))) v4i32;
typedef int __attribute__((ext_vector_type(8))) v8i32;

#if __has_include(<hip/amd_detail/amd_gfx1250_TDM.h>)
#define TDM_6ARG 1
#endif

__device__ inline unsigned short f2bf(float f) {
    unsigned int u = __builtin_bit_cast(unsigned int, f);
    u += 0x7FFFu + ((u >> 16) & 1u);          // round-to-nearest-even
    return (unsigned short)(u >> 16);
}
__device__ inline float bf2f(unsigned short h) {
    unsigned int u = ((unsigned int)h) << 16;
    return __builtin_bit_cast(float, u);
}

// A fragment (16x32 bf16): lanes 0-15: M=lane, K 0..7 / 16..23; lanes 16-31: M=lane-16, K 8..15 / 24..31
__device__ inline v16bf load_a_frag(const unsigned short* row, int k0, int half) {
    v16bf r;
    ((v8bf*)&r)[0] = *(const v8bf*)(row + k0 + half * 8);
    ((v8bf*)&r)[1] = *(const v8bf*)(row + k0 + 16 + half * 8);
    return r;
}
// B fragment (32x16 bf16): lanes 0-15: N=lane, K 0..7 / 16..23; lanes 16-31: N=lane-16, K 8..15 / 24..31
__device__ inline v16bf load_b_frag(const unsigned short* row, int k0, int half) {
    v16bf r;
    const unsigned short* p = row + k0 + half * 8;
    ((v8bf*)&r)[0] = *(const v8bf*)(p);
    ((v8bf*)&r)[1] = *(const v8bf*)(p + 16);
    return r;
}

// ---------------- TDM: 1-D DMA of a contiguous panel (global -> LDS), data_size=8B
__device__ inline void tdm_load_1d(const void* gptr, u32 lds_byte_off, u32 n_qwords) {
#if __has_builtin(__builtin_amdgcn_tensor_load_to_lds)
    unsigned long long ga = (unsigned long long)gptr;
    v4u32 g0;
    g0.x = 0x1u;                                             // count=1, user descriptor
    g0.y = lds_byte_off;                                     // lds_addr (bytes)
    g0.z = (u32)(ga & 0xFFFFFFFFu);                          // global_addr[31:0]
    g0.w = (u32)((ga >> 32) & 0x01FFFFFFu) | (2u << 30);     // global_addr[56:32] | type=2
    v8i32 g1;
    g1[0] = (3 << 16);                                       // wg_mask=0, data_size=3 (8B)
    g1[1] = (int)((n_qwords & 0xFFFFu) << 16);               // tensor_dim0[15:0]
    g1[2] = (int)(((n_qwords >> 16) & 0xFFFFu) | (1u << 16));// tensor_dim0[31:16], tensor_dim1=1
    g1[3] = (int)((n_qwords & 0xFFFFu) << 16);               // tile_dim0 = n_qwords
    g1[4] = 0;                                               // tile_dim1/2 unused
    g1[5] = (int)n_qwords;                                   // tensor_dim0_stride[31:0]
    g1[6] = 0;
    g1[7] = 0;
    v4i32 z4 = {0, 0, 0, 0};
#ifdef TDM_6ARG
    v8i32 z8 = {0, 0, 0, 0, 0, 0, 0, 0};
    __builtin_amdgcn_tensor_load_to_lds(g0, g1, z4, z4, z8, 0);
#else
    __builtin_amdgcn_tensor_load_to_lds(g0, g1, z4, z4, 0);
#endif
#endif
}

// ---------------- Projection: out[b,n,h] = sum_c feat[b,c,n]*W[h,c] + bias[h]
__global__ void proj_kernel(const float* __restrict__ feat, const float* __restrict__ W,
                            const float* __restrict__ bias, float* __restrict__ out, int C) {
    __shared__ float wtile[16 * 768];
    int b = blockIdx.x;
    int n = threadIdx.x;                       // 576 threads
    const float* f = feat + (size_t)b * C * NS;
    for (int hc = 0; hc < 4; ++hc) {
        for (int i = threadIdx.x; i < 16 * C; i += blockDim.x)
            wtile[i] = W[(hc * 16 + (i / C)) * C + (i % C)];
        __syncthreads();
        float acc[16];
#pragma unroll
        for (int k = 0; k < 16; ++k) acc[k] = bias[hc * 16 + k];
        for (int c = 0; c < C; ++c) {
            __builtin_prefetch(&f[(size_t)(c + 24) * NS + n], 0, 1);  // global_prefetch_b8
            float fv = f[(size_t)c * NS + n];  // coalesced across lanes (n)
#pragma unroll
            for (int k = 0; k < 16; ++k) acc[k] = fmaf(fv, wtile[k * C + c], acc[k]);
        }
        size_t ob = ((size_t)b * NS + n) * HIDD + hc * 16;
#pragma unroll
        for (int k = 0; k < 16; ++k) out[ob + k] = acc[k];
        __syncthreads();
    }
}

// ---------------- Token-axis L2 normalize + convert to bf16
__global__ void norm_kernel(const float* __restrict__ y, unsigned short* __restrict__ xbf) {
    __shared__ float red[256];
    int b = blockIdx.x, h = blockIdx.y;
    const float* col = y + (size_t)b * NS * HIDD + h;
    float p = 0.f;
    for (int n = threadIdx.x; n < NS; n += 256) { float v = col[(size_t)n * HIDD]; p += v * v; }
    red[threadIdx.x] = p;
    __syncthreads();
    for (int s = 128; s > 0; s >>= 1) {
        if (threadIdx.x < s) red[threadIdx.x] += red[threadIdx.x + s];
        __syncthreads();
    }
    float scale = 1.0f / fmaxf(sqrtf(red[0]), 1e-12f);
    unsigned short* o = xbf + (size_t)b * NS * HIDD + h;
    for (int n = threadIdx.x; n < NS; n += 256) o[(size_t)n * HIDD] = f2bf(col[(size_t)n * HIDD] * scale);
}

// ---------------- One Sinkhorn half-step: out[i] = logc - LSE_j(Ms(i,j) + addv[j])
// Two column tiles per trip: independent WMMA accumulators interleave to fill
// the bf16 WMMA->WMMA / WMMA->VALU hazard slots.
__device__ void ot_update(const unsigned short* __restrict__ Am, const unsigned short* __restrict__ Bm,
                          const float* __restrict__ sqA, const float* __restrict__ sqB,
                          const float* __restrict__ addv, float* __restrict__ outv,
                          int wave, int lane) {
    const int half = lane >> 4, l4 = lane & 15;
    for (int rt = wave; rt < NTILES; rt += 16) {
        const unsigned short* arow = Am + (rt * 16 + l4) * HIDD;
        v16bf a0 = load_a_frag(arow, 0, half);
        v16bf a1 = load_a_frag(arow, 32, half);
        float sqa[8];
#pragma unroll
        for (int k = 0; k < 8; ++k) sqa[k] = sqA[rt * 16 + half * 8 + k];
        float m[8], s[8];
#pragma unroll
        for (int k = 0; k < 8; ++k) { m[k] = -1e30f; s[k] = 0.f; }
        for (int ct = 0; ct < NTILES; ct += 2) {
            const unsigned short* brow0 = Bm + (ct * 16 + l4) * HIDD;
            const unsigned short* brow1 = Bm + ((ct + 1) * 16 + l4) * HIDD;
            v16bf b00 = load_b_frag(brow0, 0, half), b01 = load_b_frag(brow0, 32, half);
            v16bf b10 = load_b_frag(brow1, 0, half), b11 = load_b_frag(brow1, 32, half);
            v8f c0 = {}, c1 = {};
            c0 = __builtin_amdgcn_wmma_f32_16x16x32_bf16(false, a0, false, b00, (short)0, c0, false, false);
            c1 = __builtin_amdgcn_wmma_f32_16x16x32_bf16(false, a0, false, b10, (short)0, c1, false, false);
            c0 = __builtin_amdgcn_wmma_f32_16x16x32_bf16(false, a1, false, b01, (short)0, c0, false, false);
            c1 = __builtin_amdgcn_wmma_f32_16x16x32_bf16(false, a1, false, b11, (short)0, c1, false, false);
            float sqb0 = sqB[ct * 16 + l4],      av0 = addv[ct * 16 + l4];
            float sqb1 = sqB[(ct + 1) * 16 + l4], av1 = addv[(ct + 1) * 16 + l4];
#pragma unroll
            for (int k = 0; k < 8; ++k) {
                float ms0 = fmaxf(sqa[k] + sqb0 - 2.0f * c0[k], 0.f) * INVREG;
                float ms1 = fmaxf(sqa[k] + sqb1 - 2.0f * c1[k], 0.f) * INVREG;
                float x0 = ms0 + av0, x1 = ms1 + av1;
                float xm = fmaxf(x0, x1);
                float xs2 = __expf(x0 - xm) + __expf(x1 - xm);
                float mn = fmaxf(m[k], xm);
                s[k] = s[k] * __expf(m[k] - mn) + xs2 * __expf(xm - mn);
                m[k] = mn;
            }
        }
        // LSE combine across the 16 lanes of each half (each half owns 8 rows)
#pragma unroll
        for (int off = 1; off < 16; off <<= 1) {
#pragma unroll
            for (int k = 0; k < 8; ++k) {
                float mo = __shfl_xor(m[k], off);
                float so = __shfl_xor(s[k], off);
                float mn = fmaxf(m[k], mo);
                s[k] = s[k] * __expf(m[k] - mn) + so * __expf(mo - mn);
                m[k] = mn;
            }
        }
        if (l4 == 0) {
#pragma unroll
            for (int k = 0; k < 8; ++k)
                outv[rt * 16 + half * 8 + k] = LOGA - (m[k] + __logf(s[k]));
        }
    }
}

// ---------------- Final: per-lane partial of sum_ij exp(Ms+u_i+v_j) * (Ms*reg)
__device__ float ot_final(const unsigned short* __restrict__ Am, const unsigned short* __restrict__ Bm,
                          const float* __restrict__ sqA, const float* __restrict__ sqB,
                          const float* __restrict__ u, const float* __restrict__ v,
                          int wave, int lane) {
    const int half = lane >> 4, l4 = lane & 15;
    float acc = 0.f;
    for (int rt = wave; rt < NTILES; rt += 16) {
        const unsigned short* arow = Am + (rt * 16 + l4) * HIDD;
        v16bf a0 = load_a_frag(arow, 0, half);
        v16bf a1 = load_a_frag(arow, 32, half);
        float sqa[8], uk[8];
#pragma unroll
        for (int k = 0; k < 8; ++k) {
            sqa[k] = sqA[rt * 16 + half * 8 + k];
            uk[k]  = u[rt * 16 + half * 8 + k];
        }
        for (int ct = 0; ct < NTILES; ct += 2) {
            const unsigned short* brow0 = Bm + (ct * 16 + l4) * HIDD;
            const unsigned short* brow1 = Bm + ((ct + 1) * 16 + l4) * HIDD;
            v16bf b00 = load_b_frag(brow0, 0, half), b01 = load_b_frag(brow0, 32, half);
            v16bf b10 = load_b_frag(brow1, 0, half), b11 = load_b_frag(brow1, 32, half);
            v8f c0 = {}, c1 = {};
            c0 = __builtin_amdgcn_wmma_f32_16x16x32_bf16(false, a0, false, b00, (short)0, c0, false, false);
            c1 = __builtin_amdgcn_wmma_f32_16x16x32_bf16(false, a0, false, b10, (short)0, c1, false, false);
            c0 = __builtin_amdgcn_wmma_f32_16x16x32_bf16(false, a1, false, b01, (short)0, c0, false, false);
            c1 = __builtin_amdgcn_wmma_f32_16x16x32_bf16(false, a1, false, b11, (short)0, c1, false, false);
            float sqb0 = sqB[ct * 16 + l4],      vj0 = v[ct * 16 + l4];
            float sqb1 = sqB[(ct + 1) * 16 + l4], vj1 = v[(ct + 1) * 16 + l4];
#pragma unroll
            for (int k = 0; k < 8; ++k) {
                float ms0 = fmaxf(sqa[k] + sqb0 - 2.0f * c0[k], 0.f) * INVREG;
                float ms1 = fmaxf(sqa[k] + sqb1 - 2.0f * c1[k], 0.f) * INVREG;
                acc += __expf(ms0 + uk[k] + vj0) * (ms0 * REGC);
                acc += __expf(ms1 + uk[k] + vj1) * (ms1 * REGC);
            }
        }
    }
    return acc;
}

// ---------------- Fused Sinkhorn: one block per batch, everything LDS-resident
__global__ void __launch_bounds__(512) sinkhorn_kernel(const unsigned short* __restrict__ xsn,
                                                       const unsigned short* __restrict__ xtn,
                                                       float* __restrict__ partials) {
    extern __shared__ char smem[];
    unsigned short* xs = (unsigned short*)smem;                 // 576*64 bf16 @ LDS byte 0
    unsigned short* xt = xs + NS * HIDD;                        // @ LDS byte 73728
    float* sq_s = (float*)(xt + NT * HIDD);
    float* sq_t = sq_s + NS;
    float* u    = sq_t + NT;
    float* v    = u + NS;
    float* red  = v + NT;                                       // 16 floats
    int b = blockIdx.x, tid = threadIdx.x;
    int lane = tid & 31, wave = tid >> 5;

#if __has_builtin(__builtin_amdgcn_tensor_load_to_lds)
    // TDM DMA: both bf16 panels straight into LDS, tracked by TENSORcnt.
    if (wave == 0) {
        tdm_load_1d(xsn + (size_t)b * NS * HIDD, 0u,                  (u32)(NS * HIDD * 2 / 8));
        tdm_load_1d(xtn + (size_t)b * NT * HIDD, (u32)(NS * HIDD * 2), (u32)(NT * HIDD * 2 / 8));
#if __has_builtin(__builtin_amdgcn_s_wait_tensorcnt)
        __builtin_amdgcn_s_wait_tensorcnt(0);
#endif
    }
#else
    const uint4* gs = (const uint4*)(xsn + (size_t)b * NS * HIDD);
    const uint4* gt = (const uint4*)(xtn + (size_t)b * NT * HIDD);
    uint4* ls = (uint4*)xs;
    uint4* lt = (uint4*)xt;
    for (int i = tid; i < NS * HIDD / 8; i += 512) { ls[i] = gs[i]; lt[i] = gt[i]; }
#endif
    __syncthreads();

    for (int n = tid; n < NS; n += 512) {
        float a = 0.f, t2 = 0.f;
        for (int h = 0; h < HIDD; ++h) {
            float xv = bf2f(xs[n * HIDD + h]); a  += xv * xv;
            float tv = bf2f(xt[n * HIDD + h]); t2 += tv * tv;
        }
        sq_s[n] = a; sq_t[n] = t2; u[n] = 0.f; v[n] = 0.f;
    }
    __syncthreads();

    for (int it = 0; it < NITERS; ++it) {
        ot_update(xs, xt, sq_s, sq_t, v, u, wave, lane);   // u-update (reduce over j)
        __syncthreads();
        ot_update(xt, xs, sq_t, sq_s, u, v, wave, lane);   // v-update (transposed roles)
        __syncthreads();
    }

    float acc = ot_final(xs, xt, sq_s, sq_t, u, v, wave, lane);
#pragma unroll
    for (int off = 16; off > 0; off >>= 1) acc += __shfl_xor(acc, off);
    if (lane == 0) red[wave] = acc;
    __syncthreads();
    if (tid == 0) {
        float t = 0.f;
        for (int w2 = 0; w2 < 16; ++w2) t += red[w2];
        partials[b] = t;
    }
}

__global__ void finish_kernel(const float* __restrict__ partials, float* __restrict__ out) {
    float acc = partials[threadIdx.x];   // 32 threads
#pragma unroll
    for (int off = 16; off > 0; off >>= 1) acc += __shfl_xor(acc, off);
    if (threadIdx.x == 0) out[0] = acc * (1.0f / BS);
}

extern "C" void kernel_launch(void* const* d_in, const int* in_sizes, int n_in,
                              void* d_out, int out_size, void* d_ws, size_t ws_size,
                              hipStream_t stream) {
    const float* feat_s = (const float*)d_in[0];
    const float* feat_t = (const float*)d_in[1];
    const float* Ws     = (const float*)d_in[2];
    const float* bsv    = (const float*)d_in[3];
    const float* Wt     = (const float*)d_in[4];
    const float* btv    = (const float*)d_in[5];
    float* out = (float*)d_out;

    char* ws = (char*)d_ws;
    size_t off = 0;
    float* ys = (float*)(ws + off);          off += (size_t)BS * NS * HIDD * sizeof(float);
    float* yt = (float*)(ws + off);          off += (size_t)BS * NT * HIDD * sizeof(float);
    unsigned short* xsn = (unsigned short*)(ws + off); off += (size_t)BS * NS * HIDD * sizeof(unsigned short);
    unsigned short* xtn = (unsigned short*)(ws + off); off += (size_t)BS * NT * HIDD * sizeof(unsigned short);
    float* partials = (float*)(ws + off);    off += (size_t)BS * sizeof(float);

    proj_kernel<<<BS, NS, 0, stream>>>(feat_s, Ws, bsv, ys, 640);
    proj_kernel<<<BS, NS, 0, stream>>>(feat_t, Wt, btv, yt, 768);
    norm_kernel<<<dim3(BS, HIDD), 256, 0, stream>>>(ys, xsn);
    norm_kernel<<<dim3(BS, HIDD), 256, 0, stream>>>(yt, xtn);

    const int SMEM = (NS * HIDD + NT * HIDD) * 2 + (NS + NT) * 2 * 4 + 64;  // ~156.7 KB
    hipFuncSetAttribute((const void*)sinkhorn_kernel,
                        hipFuncAttributeMaxDynamicSharedMemorySize, SMEM);
    sinkhorn_kernel<<<BS, 512, SMEM, stream>>>(xsn, xtn, partials);
    finish_kernel<<<1, 32, 0, stream>>>(partials, out);
}
